// RNNModel_16286515986655
// MI455X (gfx1250) — matrix operand order repaired and checked by reading, not compiled, not run on hardware
//
#include <hip/hip_runtime.h>

#define BB 64
#define TT 512
#define DD 2048
#define LL 4
#define VV 64

typedef unsigned short u16;
typedef __attribute__((ext_vector_type(16))) __bf16    v16bf;
typedef __attribute__((ext_vector_type(8)))  float     v8f;
typedef __attribute__((ext_vector_type(4)))  unsigned  uintx4;

union FragA { v16bf v; uintx4 q[2]; };

__device__ __forceinline__ u16 f2bf(float f) {
  union { float f; unsigned u; } x; x.f = f;
  unsigned r = x.u + 0x7FFFu + ((x.u >> 16) & 1u);   // round-to-nearest-even
  return (u16)(r >> 16);
}
__device__ __forceinline__ float bf2f(u16 h) {
  union { unsigned u; float f; } x; x.u = ((unsigned)h) << 16;
  return x.f;
}

// ---------------------------------------------------------------------------
// One-time (per call) conversion: weights/emb -> bf16, Wout transposed,
// combined bias, zeroed double-buffered hidden state.
// ---------------------------------------------------------------------------
__global__ __launch_bounds__(256) void prep_kernel(
    const float* __restrict__ emb, const float* __restrict__ Wx,
    const float* __restrict__ bx,  const float* __restrict__ Wh,
    const float* __restrict__ bh,  const float* __restrict__ Wout,
    u16* __restrict__ embb, u16* __restrict__ wxb, u16* __restrict__ whb,
    float* __restrict__ bias, u16* __restrict__ woutb,
    u16* __restrict__ h0, u16* __restrict__ h1)
{
  const size_t NW = (size_t)LL * DD * DD;
  const size_t NE = (size_t)VV * DD;
  const size_t NB = (size_t)LL * DD;
  const size_t NH = (size_t)LL * BB * DD;
  const size_t total  = 2 * NW + 2 * NE + NB + NH;
  const size_t stride = (size_t)gridDim.x * blockDim.x;
  for (size_t i = (size_t)blockIdx.x * blockDim.x + threadIdx.x; i < total; i += stride) {
    size_t j = i;
    if (j < NW) { wxb[j] = f2bf(Wx[j]); continue; }
    j -= NW;
    if (j < NW) { whb[j] = f2bf(Wh[j]); continue; }
    j -= NW;
    if (j < NE) { embb[j] = f2bf(emb[j]); continue; }
    j -= NE;
    if (j < NE) { woutb[j] = f2bf(Wout[(j % DD) * VV + (j / DD)]); continue; } // WoutT[v][k]
    j -= NE;
    if (j < NB) { bias[j] = bx[j] + bh[j]; continue; }
    j -= NB;
    h0[j] = 0; h1[j] = 0;
  }
}

// ---------------------------------------------------------------------------
// One timestep.
//  Blocks 0..511: one (layer, 16-wide N-tile) per block; the block's 4 waves
//    split K 4 ways (512 each), accumulate all 4 batch M-tiles, reduce the
//    partial C fragments through LDS, then wave w finalizes M-tile w
//    (tanh + seq-length mask, bf16 state store).
//  Blocks 512..515 (16 waves): projection of timestep t-1 from hcur[L-1].
//  H layout: (L, B, D) bf16, double-buffered across launches.
// ---------------------------------------------------------------------------
__global__ __launch_bounds__(128) void rnn_step_kernel(
    const int* __restrict__ ids, const int* __restrict__ seq,
    const u16* __restrict__ embb, const u16* __restrict__ wxb,
    const u16* __restrict__ whb,  const float* __restrict__ bias,
    const u16* __restrict__ woutb, const float* __restrict__ bout,
    const u16* __restrict__ hcur, u16* __restrict__ hnext,
    float* __restrict__ out, int t)
{
  // partial C fragments: [k-slice wave][m-tile][lane][8 floats] = 16 KB
  __shared__ float red[4 * 4 * 32 * 8];

  const int lane  = threadIdx.x & 31;
  const int wid   = threadIdx.x >> 5;
  const int lm    = lane & 15;
  const int lhalf = lane >> 4;       // 0: lanes 0-15, 1: lanes 16-31
  const int koffA = lhalf * 16;      // A-fragment K sub-offset (halves)

  if (blockIdx.x < 512) {
    if (t >= TT) return;                       // final launch: projection only
    const int l     = blockIdx.x >> 7;         // layer 0..3
    const int nbase = (blockIdx.x & 127) * 16; // output-feature tile
    const int k0    = wid * (DD / 4);          // this wave's K slice [k0, k0+512)

    // Per-lane A-operand row base pointers (lane holds matrix row M = lm).
    const u16* axb[4];
    const u16* ahb[4];
#pragma unroll
    for (int mt = 0; mt < 4; ++mt) {
      const int b = mt * 16 + lm;
      axb[mt] = (l == 0) ? (embb + (size_t)ids[b * TT + t] * DD)          // fused embedding gather
                         : (hcur + ((size_t)(l - 1) * BB + b) * DD);      // staircase input
      ahb[mt] = hcur + ((size_t)l * BB + b) * DD;
    }
    // B-operand: lane n holds weight row (nbase+n); lhalf selects K 0-15 / 16-31.
    const u16* bxr = wxb + ((size_t)l * DD + (nbase + lm)) * (size_t)DD + lhalf * 16;
    const u16* bhr = whb + ((size_t)l * DD + (nbase + lm)) * (size_t)DD + lhalf * 16;

    v8f acc[4] = {};
    for (int k = k0; k < k0 + DD / 4; k += 32) {
      v16bf fbx = *(const v16bf*)(bxr + k);    // 32B contiguous -> 2x b128
      v16bf fbh = *(const v16bf*)(bhr + k);
      __builtin_prefetch((const void*)(bxr + k + 256), 0, 0);
      __builtin_prefetch((const void*)(bhr + k + 256), 0, 0);
#pragma unroll
      for (int mt = 0; mt < 4; ++mt) {
        FragA ax, ah;
        ax.q[0] = *(const uintx4*)(axb[mt] + k + koffA);
        ax.q[1] = *(const uintx4*)(axb[mt] + k + koffA + 8);
        ah.q[0] = *(const uintx4*)(ahb[mt] + k + koffA);
        ah.q[1] = *(const uintx4*)(ahb[mt] + k + koffA + 8);
        acc[mt] = __builtin_amdgcn_wmma_f32_16x16x32_bf16(
            false, ax.v, false, fbx, (short)0, acc[mt], false, false);
        acc[mt] = __builtin_amdgcn_wmma_f32_16x16x32_bf16(
            false, ah.v, false, fbh, (short)0, acc[mt], false, false);
      }
    }

    // Deposit partial fragments to LDS (32B-aligned v8f stores -> ds_store_b128s).
#pragma unroll
    for (int mt = 0; mt < 4; ++mt)
      *(v8f*)(&red[(((size_t)wid * 4 + mt) * 32 + lane) * 8]) = acc[mt];
    __syncthreads();

    // Wave `wid` reduces and finalizes M-tile mt = wid.
    v8f tot = {};
#pragma unroll
    for (int sw = 0; sw < 4; ++sw)
      tot += *(const v8f*)(&red[(((size_t)sw * 4 + wid) * 32 + lane) * 8]);

    const int   n  = nbase + lm;               // each lane owns one column N
    const float bv = bias[l * DD + n];
#pragma unroll
    for (int i = 0; i < 8; ++i) {
      const int    b    = wid * 16 + lhalf * 8 + i;   // C/D layout: M = i (+8 for hi lanes)
      const size_t hidx = ((size_t)l * BB + b) * (size_t)DD + n;
      float v;
      if (t < seq[b]) v = tanhf(tot[i] + bv);
      else            v = bf2f(hcur[hidx]);           // masked: keep old state
      hnext[hidx] = f2bf(v);
    }
  } else {
    // -------- projection of timestep t-1: out[:, t-1, :] = top @ Wout + bout
    if (t == 0) return;
    const int w  = (blockIdx.x - 512) * 4 + wid;   // 0..15
    const int mt = w >> 2;
    const int vt = w & 3;
    const u16* ab = hcur + ((size_t)(LL - 1) * BB + (mt * 16 + lm)) * (size_t)DD;
    const u16* br = woutb + (size_t)(vt * 16 + lm) * (size_t)DD + lhalf * 16;
    v8f acc = {};
    for (int k = 0; k < DD; k += 32) {
      v16bf fb = *(const v16bf*)(br + k);
      FragA fa;
      fa.q[0] = *(const uintx4*)(ab + k + koffA);
      fa.q[1] = *(const uintx4*)(ab + k + koffA + 8);
      acc = __builtin_amdgcn_wmma_f32_16x16x32_bf16(
          false, fa.v, false, fb, (short)0, acc, false, false);
    }
    const int   v  = vt * 16 + lm;
    const float bo = bout[v];
#pragma unroll
    for (int i = 0; i < 8; ++i) {
      const int b = mt * 16 + lhalf * 8 + i;
      out[((size_t)b * TT + (size_t)(t - 1)) * VV + v] = acc[i] + bo;
    }
  }
}

// ---------------------------------------------------------------------------
extern "C" void kernel_launch(void* const* d_in, const int* in_sizes, int n_in,
                              void* d_out, int out_size, void* d_ws, size_t ws_size,
                              hipStream_t stream) {
  (void)in_sizes; (void)n_in; (void)out_size; (void)ws_size;
  const int*   ids  = (const int*)d_in[0];
  const int*   seq  = (const int*)d_in[1];
  const float* emb  = (const float*)d_in[2];
  const float* Wx   = (const float*)d_in[3];
  const float* bx   = (const float*)d_in[4];
  const float* Wh   = (const float*)d_in[5];
  const float* bh   = (const float*)d_in[6];
  const float* Wout = (const float*)d_in[7];
  const float* bout = (const float*)d_in[8];
  float* out = (float*)d_out;

  char* p = (char*)d_ws;
  auto take = [&](size_t bytes) {
    char* r = p; p += (bytes + 255) & ~(size_t)255; return r;
  };
  u16*   wxb   = (u16*)take(sizeof(u16) * (size_t)LL * DD * DD);   // 33.5 MB
  u16*   whb   = (u16*)take(sizeof(u16) * (size_t)LL * DD * DD);   // 33.5 MB
  u16*   embb  = (u16*)take(sizeof(u16) * (size_t)VV * DD);
  u16*   woutb = (u16*)take(sizeof(u16) * (size_t)VV * DD);
  float* bias  = (float*)take(sizeof(float) * (size_t)LL * DD);
  u16*   h0    = (u16*)take(sizeof(u16) * (size_t)LL * BB * DD);
  u16*   h1    = (u16*)take(sizeof(u16) * (size_t)LL * BB * DD);

  prep_kernel<<<2048, 256, 0, stream>>>(emb, Wx, bx, Wh, bh, Wout,
                                        embb, wxb, whb, bias, woutb, h0, h1);

  // 513 launches: launch t runs RNN step t (t<T) and projection of step t-1 (t>0).
  for (int t = 0; t <= TT; ++t) {
    const u16* hc = (t & 1) ? h1 : h0;
    u16*       hn = (t & 1) ? h0 : h1;
    rnn_step_kernel<<<516, 128, 0, stream>>>(ids, seq, embb, wxb, whb, bias,
                                             woutb, bout, hc, hn, out, t);
  }
}
